// mLSTMCell_56135222558784
// MI455X (gfx1250) — matrix operand order repaired and checked by reading, not compiled, not run on hardware
//
#include <hip/hip_runtime.h>

#define B_ 128
#define D_ 1024
#define U_ 1024

typedef __attribute__((ext_vector_type(16))) __bf16 v16bf;
typedef __attribute__((ext_vector_type(8)))  float  v8f;
typedef __attribute__((ext_vector_type(4)))  float  f4;
typedef __attribute__((ext_vector_type(2)))  float  f2;

// ---------------------------------------------------------------------------
// Projection GEMMs: out[z] = X @ W[z] + bias[z]   (z = k, v, q)
// One wave (32 lanes, wave32) computes one 16x16 tile of the [B,U] output via
// v_wmma_f32_16x16x32_bf16, stepping K by 32. VGPR fragment layouts follow
// CDNA5 ISA 7.12.2 (A: 16x32 bf16; B: 32x16 bf16; C/D: 16x16 f32).
// ---------------------------------------------------------------------------
__global__ __launch_bounds__(32) void proj_wmma_kernel(
    const float* __restrict__ X,
    const float* __restrict__ Wk, const float* __restrict__ bk,
    const float* __restrict__ Wv, const float* __restrict__ bv,
    const float* __restrict__ Wq, const float* __restrict__ bq,
    float* __restrict__ ws) {
  const float* W; const float* bias; float* out;
  const int z = blockIdx.z;
  if (z == 0)      { W = Wk; bias = bk; out = ws; }
  else if (z == 1) { W = Wv; bias = bv; out = ws + B_ * U_; }
  else             { W = Wq; bias = bq; out = ws + 2 * B_ * U_; }

  const int lane = threadIdx.x;      // 0..31
  const int hl   = lane >> 4;        // lane half: 0 or 1
  const int l16  = lane & 15;
  const int M0   = blockIdx.y * 16;  // batch-tile
  const int N0   = blockIdx.x * 16;  // unit-tile

  v8f acc = {};
  const float* xr = X + (M0 + l16) * D_;  // A: row M = M0 + l16
  const float* wc = W + (N0 + l16);       // B: col N = N0 + l16 (W is [D,U] row-major)

  for (int k0 = 0; k0 < D_; k0 += 32) {
    v16bf afrag, bfrag;
    // A-frag (16x32 bf16): VGPR v holds K = 16*(v/4) + 8*half + 2*(v%4) (+1)
    #pragma unroll
    for (int v = 0; v < 8; ++v) {
      const int kk = k0 + ((v >> 2) << 4) + (hl << 3) + ((v & 3) << 1);
      f2 x2 = *(const f2*)(xr + kk);
      afrag[2 * v]     = (__bf16)x2.x;
      afrag[2 * v + 1] = (__bf16)x2.y;
    }
    // B-frag (32x16 bf16): VGPR v holds K = 16*(v/4) + 8*half + 2*(v%4) (+1)
    #pragma unroll
    for (int v = 0; v < 8; ++v) {
      const int kk = k0 + ((v >> 2) << 4) + (hl << 3) + ((v & 3) << 1);
      bfrag[2 * v]     = (__bf16)wc[(size_t)kk * U_];
      bfrag[2 * v + 1] = (__bf16)wc[(size_t)(kk + 1) * U_];
    }
    acc = __builtin_amdgcn_wmma_f32_16x16x32_bf16(
        /*neg_a=*/false, afrag, /*neg_b=*/false, bfrag,
        /*c_mod=*/(short)0, acc, /*reuse_a=*/false, /*reuse_b=*/false);
  }

  // C/D layout: VGPR r -> (M = M0 + 8*half + r, N = N0 + l16)
  const float bb = bias[N0 + l16];
  #pragma unroll
  for (int r = 0; r < 8; ++r) {
    const int m = M0 + (hl << 3) + r;
    out[m * U_ + N0 + l16] = acc[r] + bb;
  }
}

// ---------------------------------------------------------------------------
// s[b] = dot(k[b], q[b])   (one 256-thread block per batch row)
// ---------------------------------------------------------------------------
__global__ __launch_bounds__(256) void kq_dot_kernel(const float* __restrict__ ws,
                                                     float* __restrict__ s) {
  const float* k = ws;                 // k at ws + 0
  const float* q = ws + 2 * B_ * U_;   // q at ws + 2*B*U
  const int b = blockIdx.x;
  const int t = threadIdx.x;
  float acc = 0.f;
  for (int j = t; j < U_; j += 256) acc += k[b * U_ + j] * q[b * U_ + j];
  // wave32 reduce
  #pragma unroll
  for (int off = 16; off > 0; off >>= 1) acc += __shfl_down(acc, off, 32);
  __shared__ float red[8];
  if ((t & 31) == 0) red[t >> 5] = acc;
  __syncthreads();
  if (t == 0) {
    float tot = 0.f;
    #pragma unroll
    for (int w = 0; w < 8; ++w) tot += red[w];
    s[b] = tot;
  }
}

// ---------------------------------------------------------------------------
// h[b][i] = v[b][i] * s[b]; written twice (outputs 0 and 1 of the tuple)
// ---------------------------------------------------------------------------
__global__ __launch_bounds__(256) void h_kernel(const float* __restrict__ ws,
                                                const float* __restrict__ s,
                                                float* __restrict__ out) {
  const int idx = blockIdx.x * 256 + threadIdx.x;  // < B*U
  const float* v = ws + B_ * U_;
  const float h = v[idx] * s[idx >> 10];           // U_ == 1024
  out[idx] = h;
  out[B_ * U_ + idx] = h;
}

// ---------------------------------------------------------------------------
// c[b][i][j] = v[b][i] * k[b][j]  -- 537 MB streaming write, the roofline
// bound. One block per (b,i) row; float4 non-temporal stores (c_t > L2 and is
// never re-read, so bypass cache retention). k[b] row (4 KB) stays hot in L0.
// ---------------------------------------------------------------------------
__global__ __launch_bounds__(256) void outer_kernel(const float* __restrict__ ws,
                                                    float* __restrict__ outc) {
  const float* kv = ws;
  const float* vv = ws + B_ * U_;
  const int i = blockIdx.x;
  const int b = blockIdx.y;
  const float vi = vv[b * U_ + i];
  const f4* krow = (const f4*)(kv + b * U_);
  f4 kj = krow[threadIdx.x];
  f4 c;
  c.x = vi * kj.x; c.y = vi * kj.y; c.z = vi * kj.z; c.w = vi * kj.w;
  f4* crow = (f4*)(outc + ((size_t)(b * U_ + i)) * U_);
  __builtin_nontemporal_store(c, crow + threadIdx.x);
}

extern "C" void kernel_launch(void* const* d_in, const int* in_sizes, int n_in,
                              void* d_out, int out_size, void* d_ws, size_t ws_size,
                              hipStream_t stream) {
  (void)in_sizes; (void)n_in; (void)out_size; (void)ws_size;
  // setup_inputs order: inputs, h_prev, c_prev, Wk, bk, Wv, bv, Wq, bq
  const float* X  = (const float*)d_in[0];
  const float* Wk = (const float*)d_in[3];
  const float* bk = (const float*)d_in[4];
  const float* Wv = (const float*)d_in[5];
  const float* bv = (const float*)d_in[6];
  const float* Wq = (const float*)d_in[7];
  const float* bq = (const float*)d_in[8];

  float* ws  = (float*)d_ws;              // k | v | q | s  (3*B*U + B floats)
  float* s   = ws + 3 * B_ * U_;
  float* out = (float*)d_out;             // [h (B*U) | h (B*U) | c (B*U*U)]
  float* outc = out + 2 * (size_t)B_ * U_;

  dim3 pgrid(U_ / 16, B_ / 16, 3);        // 64 x 8 x 3 tiles, 1 wave each
  proj_wmma_kernel<<<pgrid, 32, 0, stream>>>(X, Wk, bk, Wv, bv, Wq, bq, ws);

  kq_dot_kernel<<<B_, 256, 0, stream>>>(ws, s);
  h_kernel<<<(B_ * U_) / 256, 256, 0, stream>>>(ws, s, out);

  dim3 ogrid(U_, B_, 1);                  // one block per (i, b) row of c_t
  outer_kernel<<<ogrid, 256, 0, stream>>>(ws, outc);
}
